// Attention_4595615007473
// MI455X (gfx1250) — matrix operand order repaired and checked by reading, not compiled, not run on hardware
//
#include <hip/hip_runtime.h>
#include <hip/hip_bf16.h>

// ---------------------------------------------------------------------------
// Types for CDNA5 WMMA (gfx1250, wave32)
// ---------------------------------------------------------------------------
typedef __attribute__((ext_vector_type(16))) __bf16 v16bf;
typedef __attribute__((ext_vector_type(8)))  float  v8f;

struct __align__(16) U4 { unsigned int x, y, z, w; };

union Frag16 {
  v16bf          v;
  unsigned short u[16];
  U4             q[2];
};

__device__ __forceinline__ v8f vzero8() {
  v8f z = {0.f, 0.f, 0.f, 0.f, 0.f, 0.f, 0.f, 0.f};
  return z;
}

__device__ __forceinline__ v8f wmma_bf16(v16bf a, v16bf b, v8f c) {
  // D = A(16x32 bf16) * B(32x16 bf16) + C(16x16 f32)
  return __builtin_amdgcn_wmma_f32_16x16x32_bf16(
      /*neg_a=*/false, a, /*neg_b=*/false, b,
      /*c_mod=*/(short)0, c, /*reuse_a=*/false, /*reuse_b=*/false);
}

__device__ __forceinline__ unsigned short f2bf(float f) {
  unsigned int u = __float_as_uint(f);
  unsigned int r = (u + 0x7FFFu + ((u >> 16) & 1u)) >> 16;
  return (unsigned short)r;
}
__device__ __forceinline__ float bf2f(unsigned short h) {
  return __uint_as_float(((unsigned int)h) << 16);
}

// Wave-local LDS ordering (CDNA5 split wait counters)
__device__ __forceinline__ void lds_fence() {
  asm volatile("s_wait_dscnt 0" ::: "memory");
}

// ---------------------------------------------------------------------------
// Problem constants
// ---------------------------------------------------------------------------
#define BATCH    2
#define SEQ      2048
#define TOKENS   (BATCH * SEQ)      // 4096
#define HIDDEN   4096
#define NHEADS   32
#define HEADDIM  128
#define N3       (3 * HIDDEN)       // 12288
#define ROPE_BASE 10000.0f

// ---------------------------------------------------------------------------
// Kernel: fp32 -> bf16 (grid-stride)
// ---------------------------------------------------------------------------
__global__ void attn_cvt_bf16(const float* __restrict__ in,
                              unsigned short* __restrict__ out, size_t n) {
  size_t i = (size_t)blockIdx.x * blockDim.x + threadIdx.x;
  size_t stride = (size_t)gridDim.x * blockDim.x;
  for (; i < n; i += stride) out[i] = f2bf(in[i]);
}

// fp32 [R,C] -> bf16 [C,R]  (so GEMM B operand is [K,N] row-major)
__global__ void attn_cvt_transpose_bf16(const float* __restrict__ in,
                                        unsigned short* __restrict__ out,
                                        int R, int C) {
  size_t i = (size_t)blockIdx.x * blockDim.x + threadIdx.x;
  if (i >= (size_t)R * C) return;
  int r = (int)(i / C);
  int c = (int)(i % C);
  out[(size_t)c * R + r] = f2bf(in[i]);
}

// ---------------------------------------------------------------------------
// WMMA GEMM: C[M,N] = A[M,K](bf16) * B[K,N](bf16), f32 accumulate.
// Block 256 threads = 8 waves; block tile 128x128, k-step 32, double-buffered
// LDS. Waves form a 4(M)x2(N) grid: each wave owns a 32x64 tile
// (2 A-frags + 4 B-frags -> 8 wmma per k-step).
// ---------------------------------------------------------------------------
#define GT_M 128
#define GT_N 128
#define GT_K 32

template <bool OUT_BF16>
__global__ __launch_bounds__(256)
void attn_gemm_wmma(const unsigned short* __restrict__ A,
                    const unsigned short* __restrict__ B,
                    void* __restrict__ C, int M, int N, int K) {
  __shared__ __align__(16) unsigned short sA[2][GT_M][GT_K];  // 2 x 8 KB
  __shared__ __align__(16) unsigned short sB[2][GT_K][GT_N];  // 2 x 8 KB

  const int tid   = threadIdx.x;
  const int lane  = tid & 31;
  const int wave  = tid >> 5;      // 0..7
  const int wm    = wave >> 1;     // 0..3 -> M offset wm*32
  const int wn    = wave & 1;      // 0..1 -> N offset wn*64
  const int l16   = lane & 15;
  const int lhalf = lane >> 4;

  const int bm = blockIdx.y * GT_M;
  const int bn = blockIdx.x * GT_N;

  v8f acc[2][4];
#pragma unroll
  for (int i = 0; i < 2; ++i)
#pragma unroll
    for (int j = 0; j < 4; ++j) acc[i][j] = vzero8();

  const int ar = tid >> 1;              // 0..127
  const int ac = (tid & 1) * 16;        // 0 / 16
  const int br = tid >> 3;              // 0..31
  const int bc = (tid & 7) * 16;        // 0..112

  const unsigned short* gA = A + (size_t)(bm + ar) * K + ac;
  const unsigned short* gB = B + (size_t)br * N + bn + bc;

  // Prologue: stage tile 0.
  {
    const U4* srcA = (const U4*)gA;
    U4* dstA = (U4*)&sA[0][ar][ac];
    dstA[0] = srcA[0];
    dstA[1] = srcA[1];
    const U4* srcB = (const U4*)gB;
    U4* dstB = (U4*)&sB[0][br][bc];
    dstB[0] = srcB[0];
    dstB[1] = srcB[1];
  }
  __syncthreads();

  int cur = 0;
  for (int k0 = 0; k0 < K; k0 += GT_K) {
    // Stage next tile into the other buffer while computing this one.
    if (k0 + GT_K < K) {
      const U4* srcA = (const U4*)(gA + k0 + GT_K);
      U4* dstA = (U4*)&sA[cur ^ 1][ar][ac];
      dstA[0] = srcA[0];
      dstA[1] = srcA[1];
      const U4* srcB = (const U4*)(gB + (size_t)(k0 + GT_K) * N);
      U4* dstB = (U4*)&sB[cur ^ 1][br][bc];
      dstB[0] = srcB[0];
      dstB[1] = srcB[1];
    }

    // A fragments: rows wm*32 + {0,16} + lane%16
    Frag16 af[2];
#pragma unroll
    for (int i = 0; i < 2; ++i) {
      const unsigned short* arow = &sA[cur][wm * 32 + i * 16 + l16][0];
      const int ko = lhalf * 8;
      af[i].q[0] = *(const U4*)(arow + ko);
      af[i].q[1] = *(const U4*)(arow + ko + 16);
    }
#pragma unroll
    for (int nt = 0; nt < 4; ++nt) {
      // B fragment: column n = lane%16, elements over k in table order
      Frag16 bf;
      const unsigned short* bcol = &sB[cur][0][wn * 64 + nt * 16 + l16];
#pragma unroll
      for (int s = 0; s < 16; ++s)
        bf.u[s] = bcol[((s >> 3) * 16 + lhalf * 8 + (s & 7)) * GT_N];
      acc[0][nt] = wmma_bf16(af[0].v, bf.v, acc[0][nt]);
      acc[1][nt] = wmma_bf16(af[1].v, bf.v, acc[1][nt]);
    }
    __syncthreads();
    cur ^= 1;
  }

  // Store: element (m,n): n = lane%16, m = base + (lane/16)*8 + r
#pragma unroll
  for (int i = 0; i < 2; ++i) {
    const int mrow = bm + wm * 32 + i * 16 + lhalf * 8;
#pragma unroll
    for (int nt = 0; nt < 4; ++nt) {
      const int ncol = bn + wn * 64 + nt * 16 + l16;
#pragma unroll
      for (int r = 0; r < 8; ++r) {
        float v = acc[i][nt][r];
        if (OUT_BF16)
          ((unsigned short*)C)[(size_t)(mrow + r) * N + ncol] = f2bf(v);
        else
          ((float*)C)[(size_t)(mrow + r) * N + ncol] = v;
      }
    }
  }
}

// ---------------------------------------------------------------------------
// RoPE (GPT-NeoX half-split) + rearrange:
//   QKV[T, 3H] -> Q [B,H,S,D] bf16
//              -> Kt [B,H,D,S] bf16   (TRANSPOSED: score B-fragment = 32B load)
//              -> V [B,H,S,D] bf16
// One thread per (token, head, d2) with d2 in [0,64): handles pair (d2, d2+64)
// ---------------------------------------------------------------------------
__global__ void attn_rope_split(const unsigned short* __restrict__ qkv,
                                const int* __restrict__ positions,
                                unsigned short* __restrict__ Qr,
                                unsigned short* __restrict__ Kt,
                                unsigned short* __restrict__ Vr) {
  size_t idx = (size_t)blockIdx.x * blockDim.x + threadIdx.x;
  const size_t total = (size_t)TOKENS * NHEADS * (HEADDIM / 2);
  if (idx >= total) return;

  const int d2 = (int)(idx % (HEADDIM / 2));
  const int h  = (int)((idx / (HEADDIM / 2)) % NHEADS);
  const int t  = (int)(idx / ((size_t)(HEADDIM / 2) * NHEADS));
  const int b  = t / SEQ;
  const int s  = t % SEQ;
  const int bh = b * NHEADS + h;

  const float pos = (float)positions[t];
  const float inv = powf(ROPE_BASE, -((float)(2 * d2)) / (float)HEADDIM);
  const float fr  = pos * inv;
  const float cs  = __cosf(fr);
  const float sn  = __sinf(fr);

  const unsigned short* base = qkv + (size_t)t * N3;
  const int qc = h * HEADDIM + d2;

  const float q1 = bf2f(base[qc]);
  const float q2 = bf2f(base[qc + 64]);
  const float k1 = bf2f(base[HIDDEN + qc]);
  const float k2 = bf2f(base[HIDDEN + qc + 64]);

  // Q, V row-major per head: [bh][s][d]
  const size_t o = ((size_t)bh * SEQ + s) * HEADDIM;
  Qr[o + d2]      = f2bf(q1 * cs - q2 * sn);
  Qr[o + 64 + d2] = f2bf(q2 * cs + q1 * sn);
  Vr[o + d2]      = base[2 * HIDDEN + qc];
  Vr[o + 64 + d2] = base[2 * HIDDEN + qc + 64];

  // K transposed per head: [bh][d][s]
  const size_t kt = (size_t)bh * HEADDIM * SEQ;
  Kt[kt + (size_t)d2 * SEQ + s]        = f2bf(k1 * cs - k2 * sn);
  Kt[kt + (size_t)(d2 + 64) * SEQ + s] = f2bf(k2 * cs + k1 * sn);
}

// ---------------------------------------------------------------------------
// Causal flash attention, WMMA bf16, per-wave 16 query rows, 64-key steps.
// Grid: (S/64, B*H). Block: 128 threads = 4 waves.
// K pre-transposed [B,H,D,S]: score B-fragments are contiguous 32B loads.
// Per 64-key step: QK^T = 4x(4 chained wmma), ONE online-softmax update,
// PV = 2x(8 wmma). Upper 32-key half skipped (wave-uniform) past causal edge.
// ---------------------------------------------------------------------------
__global__ __launch_bounds__(128)
void attn_flash_wmma(const unsigned short* __restrict__ Q,
                     const unsigned short* __restrict__ Kt,
                     const unsigned short* __restrict__ V,
                     unsigned short* __restrict__ Oattn) {
  __shared__ __align__(16) unsigned short pbuf[4][16][64];  // 8 KB

  const int bh    = blockIdx.y;           // b*32 + h
  const int b     = bh >> 5;
  const int h     = bh & 31;
  const int wave  = threadIdx.x >> 5;     // 0..3
  const int lane  = threadIdx.x & 31;
  const int l16   = lane & 15;
  const int lhalf = lane >> 4;
  const int qbase = blockIdx.x * 64 + wave * 16;

  const unsigned short* Qh = Q  + (size_t)bh * SEQ * HEADDIM;
  const unsigned short* Vh = V  + (size_t)bh * SEQ * HEADDIM;
  const unsigned short* Kh = Kt + (size_t)bh * HEADDIM * SEQ;

  // Preload Q fragments: 4 chunks of contraction-dim 32 each.
  Frag16 qf[4];
  {
    const unsigned short* qrow = Qh + (size_t)(qbase + l16) * HEADDIM;
#pragma unroll
    for (int c = 0; c < 4; ++c) {
      const int ko = c * 32 + lhalf * 8;
      qf[c].q[0] = *(const U4*)(qrow + ko);
      qf[c].q[1] = *(const U4*)(qrow + ko + 16);
    }
  }

  // Per-chunk K^T pointer for this lane: row d = c*32, key column lane%16.
  const unsigned short* krow[4];
#pragma unroll
  for (int c = 0; c < 4; ++c)
    krow[c] = Kh + (size_t)(c * 32) * SEQ + l16;

  v8f o[8];
#pragma unroll
  for (int t = 0; t < 8; ++t) o[t] = vzero8();
  float runmax[8], lsum[8];
#pragma unroll
  for (int r = 0; r < 8; ++r) { runmax[r] = -1e30f; lsum[r] = 0.f; }

  const float scale = 0.08838834764831845f;  // 1/sqrt(128)
  const int jend = qbase + 16;               // causal: keys < jend

  for (int j = 0; j < jend; j += 64) {
    const bool hi = (j + 32 < jend);  // wave-uniform: upper half needed?

    // Scores: s[g] covers keys j + g*16 .. j + g*16 + 15
    v8f s[4];
#pragma unroll
    for (int g = 0; g < 4; ++g) s[g] = vzero8();
#pragma unroll
    for (int c = 0; c < 4; ++c) {
      const unsigned short* kp = krow[c] + j;
#pragma unroll
      for (int g = 0; g < 2; ++g) {
        Frag16 bg;
#pragma unroll
        for (int s = 0; s < 16; ++s)
          bg.u[s] = kp[g * 16 + (size_t)((s >> 3) * 16 + lhalf * 8 + (s & 7)) * SEQ];
        s[g] = wmma_bf16(qf[c].v, bg.v, s[g]);
      }
    }
    if (hi) {
#pragma unroll
      for (int c = 0; c < 4; ++c) {
        const unsigned short* kp = krow[c] + j;
#pragma unroll
        for (int g = 2; g < 4; ++g) {
          Frag16 bg;
#pragma unroll
          for (int s = 0; s < 16; ++s)
            bg.u[s] = kp[g * 16 + (size_t)((s >> 3) * 16 + lhalf * 8 + (s & 7)) * SEQ];
          s[g] = wmma_bf16(qf[c].v, bg.v, s[g]);
        }
      }
    }
    if (j + 64 < jend) {  // prefetch next K^T / V rows
      __builtin_prefetch(krow[0] + j + 64, 0, 1);
      __builtin_prefetch(Vh + (size_t)(j + 64 + lane) * HEADDIM, 0, 1);
    }

    // ONE online-softmax update per 64 keys.
#pragma unroll
    for (int r = 0; r < 8; ++r) {
      const int mrow = qbase + lhalf * 8 + r;
      float a[4];
#pragma unroll
      for (int g = 0; g < 4; ++g) {
        const int key = j + g * 16 + l16;
        a[g] = (key <= mrow) ? s[g][r] * scale : -1e30f;
      }
      float mx = fmaxf(fmaxf(a[0], a[1]), fmaxf(a[2], a[3]));
#pragma unroll
      for (int off = 1; off < 16; off <<= 1)
        mx = fmaxf(mx, __shfl_xor(mx, off, 32));
      const float nm = fmaxf(runmax[r], mx);
      const float alpha = __expf(runmax[r] - nm);
      runmax[r] = nm;

      float p[4], ps = 0.f;
#pragma unroll
      for (int g = 0; g < 4; ++g) { p[g] = __expf(a[g] - nm); ps += p[g]; }
#pragma unroll
      for (int off = 1; off < 16; off <<= 1)
        ps += __shfl_xor(ps, off, 32);
      lsum[r] = lsum[r] * alpha + ps;
#pragma unroll
      for (int t = 0; t < 8; ++t) o[t][r] *= alpha;

      // P tile to LDS in (m, key) layout for A-fragment reload.
#pragma unroll
      for (int g = 0; g < 4; ++g)
        pbuf[wave][lhalf * 8 + r][g * 16 + l16] = f2bf(p[g]);
    }
    lds_fence();

    // P as two A-fragments (16 rows x 32 keys each).
    const unsigned short* prow = &pbuf[wave][l16][0];
    const int ko = lhalf * 8;
    Frag16 pf0;
    pf0.q[0] = *(const U4*)(prow + ko);
    pf0.q[1] = *(const U4*)(prow + ko + 16);

    // PV lower half: V rows j..j+31; lane%16 = dim column, elements over keys.
    const unsigned short* vrow0 = Vh + (size_t)j * HEADDIM + l16;
#pragma unroll
    for (int t = 0; t < 8; ++t) {
      Frag16 vf;
#pragma unroll
      for (int s = 0; s < 16; ++s)
        vf.u[s] = vrow0[t * 16 + ((s >> 3) * 16 + lhalf * 8 + (s & 7)) * HEADDIM];
      o[t] = wmma_bf16(pf0.v, vf.v, o[t]);
    }
    if (hi) {
      Frag16 pf1;
      pf1.q[0] = *(const U4*)(prow + 32 + ko);
      pf1.q[1] = *(const U4*)(prow + 32 + ko + 16);
      const unsigned short* vrow1 = Vh + (size_t)(j + 32) * HEADDIM + l16;
#pragma unroll
      for (int t = 0; t < 8; ++t) {
        Frag16 vf;
#pragma unroll
        for (int s = 0; s < 16; ++s)
          vf.u[s] = vrow1[t * 16 + ((s >> 3) * 16 + lhalf * 8 + (s & 7)) * HEADDIM];
        o[t] = wmma_bf16(pf1.v, vf.v, o[t]);
      }
    }
    lds_fence();  // pbuf reads complete before next iteration's stores
  }

  // Normalize and store attn output [T, HIDDEN] bf16.
#pragma unroll
  for (int r = 0; r < 8; ++r) {
    const float inv = 1.0f / lsum[r];
    const int srow = qbase + lhalf * 8 + r;
    const size_t trow = (size_t)(b * SEQ + srow) * HIDDEN;
#pragma unroll
    for (int t = 0; t < 8; ++t)
      Oattn[trow + h * HEADDIM + t * 16 + l16] = f2bf(o[t][r] * inv);
  }
}

// ---------------------------------------------------------------------------
// Launch
// ---------------------------------------------------------------------------
extern "C" void kernel_launch(void* const* d_in, const int* in_sizes, int n_in,
                              void* d_out, int out_size, void* d_ws, size_t ws_size,
                              hipStream_t stream) {
  const float* X      = (const float*)d_in[0];  // [B,S,H] fp32
  const int*   pos    = (const int*)d_in[1];    // [B,S]
  const float* Wpack  = (const float*)d_in[2];  // [3H,H] fp32
  const float* Wo     = (const float*)d_in[3];  // [H,H] fp32
  float*       out    = (float*)d_out;          // [B,S,H] fp32

  char* w = (char*)d_ws;
  // Region layout (bytes), with aliasing of dead buffers:
  const size_t WPT_OFF  = 0;                       // 96 MiB: WpT bf16 [4096,12288]
  const size_t QR_OFF   = 0;                       // aliases WpT after GEMM1
  const size_t KT_OFF   = (size_t)TOKENS * HIDDEN * 2;       // +32 MiB
  const size_t VR_OFF   = (size_t)2 * TOKENS * HIDDEN * 2;   // +64 MiB
  const size_t XB_OFF   = (size_t)HIDDEN * N3 * 2;           // 96 MiB mark
  const size_t WOT_OFF  = XB_OFF + (size_t)TOKENS * HIDDEN * 2;
  const size_t QKV_OFF  = WOT_OFF + (size_t)HIDDEN * HIDDEN * 2;
  const size_t ATTN_OFF = QKV_OFF;                 // aliases QKV after RoPE

  unsigned short* Xb   = (unsigned short*)(w + XB_OFF);
  unsigned short* WpT  = (unsigned short*)(w + WPT_OFF);
  unsigned short* WoT  = (unsigned short*)(w + WOT_OFF);
  unsigned short* QKV  = (unsigned short*)(w + QKV_OFF);
  unsigned short* Qr   = (unsigned short*)(w + QR_OFF);
  unsigned short* Kt   = (unsigned short*)(w + KT_OFF);
  unsigned short* Vr   = (unsigned short*)(w + VR_OFF);
  unsigned short* Attn = (unsigned short*)(w + ATTN_OFF);

  // 1) Convert X to bf16; transpose-convert weights so B operands are [K,N].
  {
    size_t n = (size_t)TOKENS * HIDDEN;
    attn_cvt_bf16<<<4096, 256, 0, stream>>>(X, Xb, n);
  }
  {
    size_t n = (size_t)N3 * HIDDEN;
    attn_cvt_transpose_bf16<<<(unsigned)((n + 255) / 256), 256, 0, stream>>>(
        Wpack, WpT, N3, HIDDEN);
  }
  {
    size_t n = (size_t)HIDDEN * HIDDEN;
    attn_cvt_transpose_bf16<<<(unsigned)((n + 255) / 256), 256, 0, stream>>>(
        Wo, WoT, HIDDEN, HIDDEN);
  }

  // 2) QKV = Xb @ WpT   [4096 x 12288], bf16 out
  {
    dim3 grid(N3 / GT_N, TOKENS / GT_M);
    attn_gemm_wmma<true><<<grid, 256, 0, stream>>>(Xb, WpT, QKV,
                                                   TOKENS, N3, HIDDEN);
  }

  // 3) RoPE + rearrange (Q,V row-major; K transposed per head)
  {
    size_t n = (size_t)TOKENS * NHEADS * (HEADDIM / 2);
    attn_rope_split<<<(unsigned)((n + 255) / 256), 256, 0, stream>>>(
        QKV, pos, Qr, Kt, Vr);
  }

  // 4) Causal flash attention -> Attn [4096 x 4096] bf16 (overwrites QKV head)
  {
    dim3 grid(SEQ / 64, BATCH * NHEADS);
    attn_flash_wmma<<<grid, 128, 0, stream>>>(Qr, Kt, Vr, Attn);
  }

  // 5) out = Attn @ WoT   [4096 x 4096], fp32 out
  {
    dim3 grid(HIDDEN / GT_N, TOKENS / GT_M);
    attn_gemm_wmma<false><<<grid, 256, 0, stream>>>(Attn, WoT, out,
                                                    TOKENS, HIDDEN, HIDDEN);
  }
  (void)in_sizes; (void)n_in; (void)out_size; (void)ws_size;
}
